// Qwen3NextDecoderLayer_74457553043854
// MI455X (gfx1250) — compile-verified, matrix operand order checked by reading, not run on hardware
//
#include <hip/hip_runtime.h>

// Qwen3-Next decoder layer for gfx1250 (MI455X), bf16 WMMA + fp32 accum.
// B=2 S=2048 H=2048 NH=16 NKV=4 HD=128 F=8192 RD=32

#define B_   2
#define S_   2048
#define H_   2048
#define NH_  16
#define NKV_ 4
#define HD_  128
#define F_   8192
#define MTOK (B_ * S_)

typedef __attribute__((ext_vector_type(16))) __bf16 v16bf;
typedef __attribute__((ext_vector_type(8)))  float  v8f;
typedef __attribute__((ext_vector_type(4)))  int    v4i;

#if __has_builtin(__builtin_amdgcn_global_load_async_to_lds_b128) && \
    __has_builtin(__builtin_amdgcn_s_wait_asynccnt)
#define ASYNC_LDS 1
#endif

#ifdef ASYNC_LDS
typedef __attribute__((address_space(1))) v4i gv4i;   // global int4
typedef __attribute__((address_space(3))) v4i lv4i;   // LDS int4
#endif

// 16B global -> LDS copy; async (ASYNCcnt) when the CDNA5 builtin exists.
__device__ __forceinline__ void copy_b128(const unsigned short* g,
                                          unsigned short* l) {
#ifdef ASYNC_LDS
  __builtin_amdgcn_global_load_async_to_lds_b128((gv4i*)g, (lv4i*)l, 0, 0);
#else
  *(uint4*)l = *(const uint4*)g;
#endif
}

__device__ __forceinline__ void async_wait0() {
#ifdef ASYNC_LDS
  __builtin_amdgcn_s_wait_asynccnt(0);
#endif
}

__device__ __forceinline__ unsigned short f2bf_bits(float f) {
  union { float f; unsigned u; } x; x.f = f;
  unsigned r = x.u + 0x7FFFu + ((x.u >> 16) & 1u);   // round-to-nearest-even
  return (unsigned short)(r >> 16);
}

__device__ __forceinline__ v8f wmma_bf16(v16bf a, v16bf b, v8f c) {
  return __builtin_amdgcn_wmma_f32_16x16x32_bf16(false, a, false, b,
                                                 (short)0, c, false, false);
}

// Load one 16-bit A/B fragment (16x32 MxK, or 32x16 KxN staged as [n][k]).
// Lane l: row = (l&15) (caller indexes the row), khalf = (l>>4)*8.
__device__ __forceinline__ v16bf load_frag(const unsigned short* p, int khalf) {
  union { uint4 u[2]; v16bf v; } f;
  f.u[0] = *(const uint4*)(p + khalf);
  f.u[1] = *(const uint4*)(p + 16 + khalf);
  return f.v;
}

// ---------------------------------------------------------------- RMSNorm(+1)
__global__ __launch_bounds__(256) void rmsnorm_bf16(
    const float* __restrict__ X, const float* __restrict__ Wn,
    unsigned short* __restrict__ Y, int Hdim) {
  __shared__ float red[8];
  const int row = blockIdx.x;
  const float* x = X + (size_t)row * Hdim;
  float ss = 0.f;
  for (int i = threadIdx.x; i < Hdim; i += 256) { float v = x[i]; ss += v * v; }
#pragma unroll
  for (int off = 16; off; off >>= 1) ss += __shfl_xor(ss, off, 32);
  if ((threadIdx.x & 31) == 0) red[threadIdx.x >> 5] = ss;
  __syncthreads();
  float tot = 0.f;
#pragma unroll
  for (int w = 0; w < 8; ++w) tot += red[w];
  float inv = rsqrtf(tot / (float)Hdim + 1e-6f);
  for (int i = threadIdx.x; i < Hdim; i += 256)
    Y[(size_t)row * Hdim + i] = f2bf_bits(x[i] * inv * (1.f + Wn[i]));
}

// ------------------------------------------------- generic bf16 WMMA GEMM
// C[M,N](f32) = A[M,K](bf16) * W[K,N](f32->bf16 on the fly) (+ Res)
#define BM 128
#define BN 128
#define BK 32
#define BKP 40   // padded K in halfs: 80B rows keep 16B-aligned b128 frags

// stage one (A,W) K-slice into one LDS buffer
__device__ __forceinline__ void stage_aw(
    const unsigned short* __restrict__ A, const float* __restrict__ W,
    int N, int K, int bm, int bn, int k0,
    unsigned short (*As)[BKP], unsigned short (*Bs)[BKP], int tid) {
  const int ar = tid >> 1, ac = (tid & 1) * 16;
  const unsigned short* ga = A + (size_t)(bm + ar) * K + k0 + ac;
  copy_b128(ga, &As[ar][ac]);
  copy_b128(ga + 8, &As[ar][ac + 8]);
  const int wkr = tid >> 3, wcg = (tid & 7) * 16;
  const float* wp = W + (size_t)(k0 + wkr) * N + bn + wcg;
  if (k0 + 2 * BK < K) __builtin_prefetch(wp + (size_t)2 * BK * N, 0, 1);
  const float4* src = (const float4*)wp;
#pragma unroll
  for (int j = 0; j < 4; ++j) {
    float4 f = src[j];
    Bs[wcg + j * 4 + 0][wkr] = f2bf_bits(f.x);
    Bs[wcg + j * 4 + 1][wkr] = f2bf_bits(f.y);
    Bs[wcg + j * 4 + 2][wkr] = f2bf_bits(f.z);
    Bs[wcg + j * 4 + 3][wkr] = f2bf_bits(f.w);
  }
}

template <int ADD_RES>
__global__ __launch_bounds__(256) void gemm_bf16_f32(
    const unsigned short* __restrict__ A, const float* __restrict__ W,
    float* __restrict__ C, const float* __restrict__ Res,
    int M, int N, int K) {
  __shared__ unsigned short As[2][BM][BKP];
  __shared__ unsigned short Bs[2][BN][BKP];   // W^T tile: [n][k]
  const int tid = threadIdx.x, lane = tid & 31, wid = tid >> 5;
  const int lo = lane & 15, hi = lane >> 4, khalf = hi * 8;
  const int bm = blockIdx.y * BM, bn = blockIdx.x * BN;
  const int wm = (wid >> 2) * 64, wn = (wid & 3) * 32;

  v8f acc[4][2] = {};

  const int nIter = K / BK;
  stage_aw(A, W, N, K, bm, bn, 0, As[0], Bs[0], tid);
  int buf = 0;
  for (int it = 0; it < nIter; ++it) {
    async_wait0();
    __syncthreads();
    if (it + 1 < nIter)
      stage_aw(A, W, N, K, bm, bn, (it + 1) * BK, As[buf ^ 1], Bs[buf ^ 1], tid);
    v16bf afr[4], bfr[2];
#pragma unroll
    for (int i = 0; i < 4; ++i)
      afr[i] = load_frag(&As[buf][wm + i * 16 + lo][0], khalf);
#pragma unroll
    for (int j = 0; j < 2; ++j)
      bfr[j] = load_frag(&Bs[buf][wn + j * 16 + lo][0], khalf);
#pragma unroll
    for (int i = 0; i < 4; ++i)
#pragma unroll
      for (int j = 0; j < 2; ++j)
        acc[i][j] = wmma_bf16(afr[i], bfr[j], acc[i][j]);
    buf ^= 1;
  }

#pragma unroll
  for (int i = 0; i < 4; ++i)
#pragma unroll
    for (int j = 0; j < 2; ++j)
#pragma unroll
      for (int v = 0; v < 8; ++v) {
        int row = bm + wm + i * 16 + v + 8 * hi;   // C layout: m = v + 8*(l>>4)
        int col = bn + wn + j * 16 + lo;           //           n = l&15
        size_t idx = (size_t)row * N + col;
        float val = acc[i][j][v];
        if (ADD_RES) val += Res[idx];
        C[idx] = val;
      }
}

// ------------------------------------- fused gate/up GEMM + silu*mul -> bf16
__device__ __forceinline__ void stage_w(
    const float* __restrict__ W, int N, int K, int bn, int k0,
    unsigned short (*Bs)[BKP], int tid) {
  const int wkr = tid >> 3, wcg = (tid & 7) * 16;
  const float* wp = W + (size_t)(k0 + wkr) * N + bn + wcg;
  if (k0 + 2 * BK < K) __builtin_prefetch(wp + (size_t)2 * BK * N, 0, 1);
  const float4* src = (const float4*)wp;
#pragma unroll
  for (int j = 0; j < 4; ++j) {
    float4 f = src[j];
    Bs[wcg + j * 4 + 0][wkr] = f2bf_bits(f.x);
    Bs[wcg + j * 4 + 1][wkr] = f2bf_bits(f.y);
    Bs[wcg + j * 4 + 2][wkr] = f2bf_bits(f.z);
    Bs[wcg + j * 4 + 3][wkr] = f2bf_bits(f.w);
  }
}

__global__ __launch_bounds__(256) void gemm_gateup(
    const unsigned short* __restrict__ A, const float* __restrict__ Wg,
    const float* __restrict__ Wu, unsigned short* __restrict__ Out,
    int M, int N, int K) {
  __shared__ unsigned short As[2][BM][BKP];
  __shared__ unsigned short Bg[2][BN][BKP];
  __shared__ unsigned short Bu[2][BN][BKP];
  const int tid = threadIdx.x, lane = tid & 31, wid = tid >> 5;
  const int lo = lane & 15, hi = lane >> 4, khalf = hi * 8;
  const int bm = blockIdx.y * BM, bn = blockIdx.x * BN;
  const int wm = (wid >> 2) * 64, wn = (wid & 3) * 32;

  v8f accg[4][2] = {};
  v8f accu[4][2] = {};

  const int ar = tid >> 1, ac = (tid & 1) * 16;
  const int nIter = K / BK;
  {
    const unsigned short* ga = A + (size_t)(bm + ar) * K + ac;
    copy_b128(ga, &As[0][ar][ac]);
    copy_b128(ga + 8, &As[0][ar][ac + 8]);
    stage_w(Wg, N, K, bn, 0, Bg[0], tid);
    stage_w(Wu, N, K, bn, 0, Bu[0], tid);
  }
  int buf = 0;
  for (int it = 0; it < nIter; ++it) {
    async_wait0();
    __syncthreads();
    if (it + 1 < nIter) {
      int k0 = (it + 1) * BK;
      const unsigned short* ga = A + (size_t)(bm + ar) * K + k0 + ac;
      copy_b128(ga, &As[buf ^ 1][ar][ac]);
      copy_b128(ga + 8, &As[buf ^ 1][ar][ac + 8]);
      stage_w(Wg, N, K, bn, k0, Bg[buf ^ 1], tid);
      stage_w(Wu, N, K, bn, k0, Bu[buf ^ 1], tid);
    }
    v16bf afr[4];
#pragma unroll
    for (int i = 0; i < 4; ++i)
      afr[i] = load_frag(&As[buf][wm + i * 16 + lo][0], khalf);
#pragma unroll
    for (int j = 0; j < 2; ++j) {
      v16bf bg = load_frag(&Bg[buf][wn + j * 16 + lo][0], khalf);
      v16bf bu = load_frag(&Bu[buf][wn + j * 16 + lo][0], khalf);
#pragma unroll
      for (int i = 0; i < 4; ++i) {
        accg[i][j] = wmma_bf16(afr[i], bg, accg[i][j]);
        accu[i][j] = wmma_bf16(afr[i], bu, accu[i][j]);
      }
    }
    buf ^= 1;
  }

#pragma unroll
  for (int i = 0; i < 4; ++i)
#pragma unroll
    for (int j = 0; j < 2; ++j)
#pragma unroll
      for (int v = 0; v < 8; ++v) {
        int row = bm + wm + i * 16 + v + 8 * hi;
        int col = bn + wn + j * 16 + lo;
        float g = accg[i][j][v], u = accu[i][j][v];
        float sig = __builtin_amdgcn_rcpf(1.f + __expf(-g));  // fast sigmoid
        Out[(size_t)row * N + col] = f2bf_bits(g * sig * u);
      }
}

// --------------------------------------- per-head RMSNorm(+1) + partial RoPE
// one wave per (token, head); lane l owns dims {l, l+32, l+64, l+96}; RoPE
// touches only dims < 32 (= lane index), pairing via shfl_xor(…,16).
__global__ __launch_bounds__(256) void qk_norm_rope(
    const float* __restrict__ src, const float* __restrict__ nw,
    const int* __restrict__ pos, unsigned short* __restrict__ dst,
    int nheads, int row_stride, int head_stride, int dst_row_stride,
    int total_waves) {
  int wg = blockIdx.x * 8 + (threadIdx.x >> 5);
  if (wg >= total_waves) return;
  int lane = threadIdx.x & 31;
  int token = wg / nheads, head = wg % nheads;
  const float* x = src + (size_t)token * row_stride + head * head_stride;
  float v0 = x[lane], v1 = x[lane + 32], v2 = x[lane + 64], v3 = x[lane + 96];
  float ss = v0 * v0 + v1 * v1 + v2 * v2 + v3 * v3;
#pragma unroll
  for (int off = 16; off; off >>= 1) ss += __shfl_xor(ss, off, 32);
  float inv = rsqrtf(ss * (1.f / 128.f) + 1e-6f);
  v0 *= inv * (1.f + nw[lane]);
  v1 *= inv * (1.f + nw[lane + 32]);
  v2 *= inv * (1.f + nw[lane + 64]);
  v3 *= inv * (1.f + nw[lane + 96]);
  // RoPE, RD=32: inv_freq[i] = theta^(-i/16), angle = pos * inv_freq[l%16]
  float p = (float)pos[token];
  float a = p * __powf(10000.f, -(float)(lane & 15) * (1.f / 16.f));
  float c = __cosf(a), s = __sinf(a);
  float other = __shfl_xor(v0, 16, 32);
  float roped = (lane < 16) ? (v0 * c - other * s) : (v0 * c + other * s);
  unsigned short* d = dst + (size_t)token * dst_row_stride + head * HD_;
  d[lane] = f2bf_bits(roped);
  d[lane + 32] = f2bf_bits(v1);
  d[lane + 64] = f2bf_bits(v2);
  d[lane + 96] = f2bf_bits(v3);
}

__global__ void f32_to_bf16(const float* __restrict__ x,
                            unsigned short* __restrict__ y, int n) {
  for (int i = blockIdx.x * 256 + threadIdx.x; i < n; i += gridDim.x * 256)
    y[i] = f2bf_bits(x[i]);
}

// ------------------------------------------------ flash attention (causal, GQA)
// grid = (S/128, NH, B); 8 waves, wave w owns query rows qbase+16w .. +15.
#define BQ  128
#define BKV 32
#define HDP 136   // K-tile row pad (halfs)
#define KVP 40    // V^T / P row pad (halfs)

__global__ __launch_bounds__(256) void attn_fwd(
    const unsigned short* __restrict__ Qb,   // [B*S][NH*HD]
    const unsigned short* __restrict__ Kb,   // [B*S][NKV*HD]
    const unsigned short* __restrict__ Vb,   // [B*S][NKV*HD]
    const float* __restrict__ qg,            // gates at [..][head*256+128+hd]
    unsigned short* __restrict__ Ob,         // [B*S][NH*HD]
    int S) {
  __shared__ unsigned short Ks[BKV][HDP];    // [key][hd]  (= B^T for Q*K^T)
  __shared__ unsigned short Vt[HD_][KVP];    // [hd][key]  (= B^T for P*V)
  __shared__ unsigned short Ps[8][16][KVP];  // per-wave P in A layout [m][k]

  const int qtile = blockIdx.x, head = blockIdx.y, batch = blockIdx.z;
  const int kvh = head >> 2;                 // NH/NKV = 4
  const int qbase = qtile * BQ;
  const int tid = threadIdx.x, lane = tid & 31, wid = tid >> 5;
  const int lo = lane & 15, hi = lane >> 4, khalf = hi * 8;
  const size_t tok0 = (size_t)batch * S;

  // Q fragments straight from global: lane = query row, k = hd
  v16bf qf[4];
  {
    const unsigned short* qp =
        Qb + (tok0 + qbase + wid * 16 + lo) * (NH_ * HD_) + head * HD_;
#pragma unroll
    for (int j = 0; j < 4; ++j) qf[j] = load_frag(qp + j * 32, khalf);
  }

  float rmax[8], rsum[8];
#pragma unroll
  for (int v = 0; v < 8; ++v) { rmax[v] = -3.0e38f; rsum[v] = 0.f; }
  v8f oacc[8] = {};

  const int nkv = (qbase + BQ) / BKV;
  const int ldr = tid >> 3, ldc = (tid & 7) * 16;

  for (int kt = 0; kt < nkv; ++kt) {
    const int kb = kt * BKV;
    __syncthreads();
    {  // K tile: [key][hd] row-major, async global->LDS (ASYNCcnt)
      const unsigned short* src =
          Kb + (tok0 + kb + ldr) * (NKV_ * HD_) + kvh * HD_ + ldc;
      copy_b128(src, &Ks[ldr][ldc]);
      copy_b128(src + 8, &Ks[ldr][ldc + 8]);
    }
    {  // V tile transposed: Vt[hd][key]
      union { uint4 u[2]; unsigned short s[16]; } t;
      const uint4* src =
          (const uint4*)(Vb + (tok0 + kb + ldr) * (NKV_ * HD_) + kvh * HD_ + ldc);
      t.u[0] = src[0]; t.u[1] = src[1];
#pragma unroll
      for (int j = 0; j < 16; ++j) Vt[ldc + j][ldr] = t.s[j];
    }
    async_wait0();
    __syncthreads();

    // scores S = Q * K^T  (16 x 32 per wave)
    v8f sc[2] = {};
#pragma unroll
    for (int nt = 0; nt < 2; ++nt)
#pragma unroll
      for (int j = 0; j < 4; ++j) {
        v16bf kf = load_frag(&Ks[nt * 16 + lo][j * 32], khalf);
        sc[nt] = wmma_bf16(qf[j], kf, sc[nt]);
      }

    const float sscale = 0.08838834764831845f;  // 1/sqrt(128)
#pragma unroll
    for (int nt = 0; nt < 2; ++nt) {
      int key = kb + nt * 16 + lo;
#pragma unroll
      for (int v = 0; v < 8; ++v) {
        int qrow = qbase + wid * 16 + v + 8 * hi;
        float s = sc[nt][v] * sscale;
        sc[nt][v] = (key <= qrow) ? s : -3.0e38f;
      }
    }

    // online softmax: row reductions across the 16 lanes that share m
    float corr[8];
#pragma unroll
    for (int v = 0; v < 8; ++v) {
      float m = fmaxf(sc[0][v], sc[1][v]);
#pragma unroll
      for (int off = 8; off; off >>= 1) m = fmaxf(m, __shfl_xor(m, off, 32));
      float nmax = fmaxf(rmax[v], m);
      corr[v] = __expf(rmax[v] - nmax);
      rmax[v] = nmax;
      float p0 = (sc[0][v] < -1.0e30f) ? 0.f : __expf(sc[0][v] - nmax);
      float p1 = (sc[1][v] < -1.0e30f) ? 0.f : __expf(sc[1][v] - nmax);
      Ps[wid][v + 8 * hi][lo] = f2bf_bits(p0);
      Ps[wid][v + 8 * hi][16 + lo] = f2bf_bits(p1);
      float ps = p0 + p1;
#pragma unroll
      for (int off = 8; off; off >>= 1) ps += __shfl_xor(ps, off, 32);
      rsum[v] = rsum[v] * corr[v] + ps;
    }
#pragma unroll
    for (int n8 = 0; n8 < 8; ++n8)
#pragma unroll
      for (int v = 0; v < 8; ++v) oacc[n8][v] *= corr[v];

    __syncthreads();  // P visible (and V^T stable) before P*V

    v16bf pf = load_frag(&Ps[wid][lo][0], khalf);  // A frag: m = lo, k = key
#pragma unroll
    for (int n8 = 0; n8 < 8; ++n8) {
      v16bf vf = load_frag(&Vt[n8 * 16 + lo][0], khalf);  // B: n = hd, k = key
      oacc[n8] = wmma_bf16(pf, vf, oacc[n8]);
    }
  }

  // epilogue: 1/l (fast rcp), sigmoid(gate), store bf16
#pragma unroll
  for (int v = 0; v < 8; ++v) rsum[v] = __builtin_amdgcn_rcpf(rsum[v]);
#pragma unroll
  for (int n8 = 0; n8 < 8; ++n8)
#pragma unroll
    for (int v = 0; v < 8; ++v) {
      int qrow = qbase + wid * 16 + v + 8 * hi;
      int hd = n8 * 16 + lo;
      float g = qg[(tok0 + qrow) * (NH_ * 2 * HD_) + head * (2 * HD_) + HD_ + hd];
      float sig = __builtin_amdgcn_rcpf(1.f + __expf(-g));
      float o = oacc[n8][v] * rsum[v] * sig;
      Ob[(tok0 + qrow) * (NH_ * HD_) + head * HD_ + hd] = f2bf_bits(o);
    }
}

// --------------------------------------------------------------- launcher
extern "C" void kernel_launch(void* const* d_in, const int* in_sizes, int n_in,
                              void* d_out, int out_size, void* d_ws,
                              size_t ws_size, hipStream_t stream) {
  const float* hidden = (const float*)d_in[0];
  const int*   pos    = (const int*)d_in[1];
  const float* ln1_w  = (const float*)d_in[2];
  const float* q_w    = (const float*)d_in[3];
  const float* k_w    = (const float*)d_in[4];
  const float* v_w    = (const float*)d_in[5];
  const float* qn_w   = (const float*)d_in[6];
  const float* kn_w   = (const float*)d_in[7];
  const float* o_w    = (const float*)d_in[8];
  const float* ln2_w  = (const float*)d_in[9];
  const float* gate_w = (const float*)d_in[10];
  const float* up_w   = (const float*)d_in[11];
  const float* down_w = (const float*)d_in[12];
  float* out = (float*)d_out;

  char* ws = (char*)d_ws;
  size_t off = 0;
  auto alloc = [&](size_t bytes) -> void* {
    void* p = ws + off;
    off = (off + bytes + 255) & ~(size_t)255;
    return p;
  };
  unsigned short* hn1  = (unsigned short*)alloc((size_t)MTOK * H_ * 2);
  float*          qgb  = (float*)alloc((size_t)MTOK * NH_ * 2 * HD_ * 4);
  float*          kbuf = (float*)alloc((size_t)MTOK * NKV_ * HD_ * 4);
  float*          vbuf = (float*)alloc((size_t)MTOK * NKV_ * HD_ * 4);
  unsigned short* qbf  = (unsigned short*)alloc((size_t)MTOK * NH_ * HD_ * 2);
  unsigned short* kbf  = (unsigned short*)alloc((size_t)MTOK * NKV_ * HD_ * 2);
  unsigned short* vbf  = (unsigned short*)alloc((size_t)MTOK * NKV_ * HD_ * 2);
  unsigned short* attn = (unsigned short*)alloc((size_t)MTOK * NH_ * HD_ * 2);
  float*          h2   = (float*)alloc((size_t)MTOK * H_ * 4);
  unsigned short* hn2  = (unsigned short*)alloc((size_t)MTOK * H_ * 2);
  unsigned short* act  = (unsigned short*)alloc((size_t)MTOK * F_ * 2);

  rmsnorm_bf16<<<MTOK, 256, 0, stream>>>(hidden, ln1_w, hn1, H_);

  gemm_bf16_f32<0><<<dim3((NH_ * 2 * HD_) / BN, MTOK / BM), 256, 0, stream>>>(
      hn1, q_w, qgb, nullptr, MTOK, NH_ * 2 * HD_, H_);
  gemm_bf16_f32<0><<<dim3((NKV_ * HD_) / BN, MTOK / BM), 256, 0, stream>>>(
      hn1, k_w, kbuf, nullptr, MTOK, NKV_ * HD_, H_);
  gemm_bf16_f32<0><<<dim3((NKV_ * HD_) / BN, MTOK / BM), 256, 0, stream>>>(
      hn1, v_w, vbuf, nullptr, MTOK, NKV_ * HD_, H_);

  qk_norm_rope<<<(MTOK * NH_) / 8, 256, 0, stream>>>(
      qgb, qn_w, pos, qbf, NH_, NH_ * 2 * HD_, 2 * HD_, NH_ * HD_, MTOK * NH_);
  qk_norm_rope<<<(MTOK * NKV_) / 8, 256, 0, stream>>>(
      kbuf, kn_w, pos, kbf, NKV_, NKV_ * HD_, HD_, NKV_ * HD_, MTOK * NKV_);
  f32_to_bf16<<<1024, 256, 0, stream>>>(vbuf, vbf, MTOK * NKV_ * HD_);

  attn_fwd<<<dim3(S_ / BQ, NH_, B_), 256, 0, stream>>>(qbf, kbf, vbf, qgb,
                                                       attn, S_);

  gemm_bf16_f32<1><<<dim3(H_ / BN, MTOK / BM), 256, 0, stream>>>(
      attn, o_w, h2, hidden, MTOK, H_, NH_ * HD_);

  rmsnorm_bf16<<<MTOK, 256, 0, stream>>>(h2, ln2_w, hn2, H_);

  gemm_gateup<<<dim3(F_ / BN, MTOK / BM), 256, 0, stream>>>(
      hn2, gate_w, up_w, act, MTOK, F_, H_);

  gemm_bf16_f32<1><<<dim3(H_ / BN, MTOK / BM), 256, 0, stream>>>(
      act, down_w, out, h2, MTOK, H_, F_);
}
